// Sequence2_47957604827672
// MI455X (gfx1250) — matrix operand order repaired and checked by reading, not compiled, not run on hardware
//
#include <hip/hip_runtime.h>

// ---------------------------------------------------------------------------
// MI455X (gfx1250) 2-layer LSTM scan.
//  - bf16 WMMA (v_wmma_f32_16x16x32_bf16), f32 accumulate
//  - gates GEMM + LSTM cell fused per layer (4 gate tiles per wave, cell in regs)
//  - activations staged to LDS via Tensor Data Mover (tensor_load_to_lds)
//  - weights pre-packed to WMMA B-fragment lane layout (L2-resident, ~25MB bf16)
// ---------------------------------------------------------------------------

typedef float  v8f   __attribute__((ext_vector_type(8)));
typedef __bf16 v16bf __attribute__((ext_vector_type(16)));
typedef unsigned int v4u __attribute__((ext_vector_type(4)));
typedef int          v8i __attribute__((ext_vector_type(8)));
typedef int          v4i __attribute__((ext_vector_type(4)));

struct __align__(16) Q16 { unsigned a, b, c, d; };
union Frag { Q16 q[2]; v16bf v; };

__device__ __forceinline__ unsigned short f2bf(float f) {
    unsigned u = __float_as_uint(f);
    unsigned r = u + 0x7FFFu + ((u >> 16) & 1u);   // round-to-nearest-even
    return (unsigned short)(r >> 16);
}
__device__ __forceinline__ float sigm(float x) { return 1.0f / (1.0f + __expf(-x)); }

// ---------------------------------------------------------------------------
// Pack [Wi | Wh] (row n = output unit, col k = input dim, K-concatenated) into
// bf16 WMMA B-fragments. Layout per (ktile,ntile): lane l holds column
// n = ntile*16 + (l&15), K values k = ktile*32 + (l>=16?8:0) + (s&7) + (s>=8?16:0).
// ---------------------------------------------------------------------------
__global__ void pack_w_kernel(const float* __restrict__ Wi, const float* __restrict__ Wh,
                              int Kin, int Kh, int NTILES,
                              unsigned short* __restrict__ out, long total) {
    long idx = (long)blockIdx.x * blockDim.x + threadIdx.x;
    if (idx >= total) return;
    int  s    = (int)(idx & 15);
    int  lane = (int)((idx >> 4) & 31);
    long rest = idx >> 9;
    int  ntile = (int)(rest % NTILES);
    int  ktile = (int)(rest / NTILES);
    int  n = ntile * 16 + (lane & 15);
    int  k = ktile * 32 + ((lane >> 4) << 3) + (s & 7) + ((s >> 3) << 4);
    float v = (k < Kin) ? Wi[(long)n * Kin + k] : Wh[(long)n * Kh + (k - Kin)];
    out[idx] = f2bf(v);
}

__global__ void bias_sum_kernel(const float* __restrict__ a, const float* __restrict__ b,
                                float* __restrict__ o, int n) {
    int i = blockIdx.x * blockDim.x + threadIdx.x;
    if (i < n) o[i] = a[i] + b[i];
}

// x_t (f32) -> bf16 into columns [0,64) of hcat1 read-buffer
__global__ void prep_x_kernel(const float* __restrict__ in, unsigned short* __restrict__ hcat1,
                              int t, int T) {
    int i = blockIdx.x * blockDim.x + threadIdx.x;   // 256*64 threads
    int b = i >> 6, k = i & 63;
    hcat1[b * 1088 + k] = f2bf(in[((long)b * T + t) * 64 + k]);
}

// future phase: x = out_{t-1} @ W_o2i^T + b_o2i  -> bf16 cols [0,64) of hcat1
__global__ void proj_in_kernel(const float* __restrict__ dout, const float* __restrict__ Wo2i,
                               const float* __restrict__ bo2i, unsigned short* __restrict__ hcat1,
                               int tprev, int TT) {
    int i = blockIdx.x * blockDim.x + threadIdx.x;   // 256*64 threads
    int b = i >> 6, j = i & 63;
    const float* o = dout + ((long)b * TT + tprev) * 64;
    float s = bo2i[j];
    #pragma unroll 4
    for (int n = 0; n < 64; ++n) s += o[n] * Wo2i[j * 64 + n];
    hcat1[b * 1088 + j] = f2bf(s);
}

// ---------------------------------------------------------------------------
// Fused gates-GEMM + LSTM cell.
//  grid = (B/16, 1024/128), block = 256 (8 waves). Each wave: one 16-row x
//  16-col hidden tile, 4 gate accumulators. A tile (16 x KCAT bf16) staged to
//  LDS once per block via TDM.
// ---------------------------------------------------------------------------
__global__ void __launch_bounds__(256)
lstm_step_kernel(const unsigned short* __restrict__ hcat, int KCAT,
                 const unsigned short* __restrict__ wpack,
                 const float* __restrict__ bias,
                 float* __restrict__ c,
                 unsigned short* __restrict__ dstA, int strideA, int offA,
                 unsigned short* __restrict__ dstB, int strideB, int offB) {
    extern __shared__ unsigned short lds[];
    const int lane = threadIdx.x & 31;
    const int wave = threadIdx.x >> 5;
    const int m0   = blockIdx.x * 16;
    const int n0   = blockIdx.y * 128 + wave * 16;

#if __has_builtin(__builtin_amdgcn_tensor_load_to_lds)
    // --- Tensor Data Mover: DMA 16 rows x KCAT bf16 of A into LDS (wave 0) ---
    if (threadIdx.x < 32) {
        unsigned long long ga = (unsigned long long)(size_t)(const void*)(hcat + (long)m0 * KCAT);
        unsigned ldsOff = (unsigned)(size_t)(const void*)lds;
        unsigned dim0   = (unsigned)(KCAT >> 1);           // row length in dwords (data_size=4B)
        v4u g0 = { 1u,                                     // count=1, user mode
                   ldsOff,                                 // lds_addr (bytes)
                   (unsigned)ga,
                   (unsigned)((ga >> 32) & 0x1FFFFFFull) | (2u << 30) };  // addr[56:32] | type=2
        v8i g1 = { (int)(2u << 16),                        // data_size = 4 bytes
                   (int)((dim0 & 0xFFFFu) << 16),          // tensor_dim0[15:0]
                   (int)((dim0 >> 16) | (16u << 16)),      // tensor_dim0[31:16] | tensor_dim1=16
                   (int)(dim0 << 16),                      // tile_dim0 = dim0
                   16,                                     // tile_dim1 = 16 rows
                   (int)dim0,                              // tensor_dim0_stride (dwords)
                   0, 0 };
        v4i g2 = {0, 0, 0, 0}, g3 = {0, 0, 0, 0};
#if defined(__clang_major__) && __clang_major__ >= 23
        v8i g4 = {0, 0, 0, 0, 0, 0, 0, 0};
        __builtin_amdgcn_tensor_load_to_lds(g0, g1, g2, g3, g4, 0);
#else
        __builtin_amdgcn_tensor_load_to_lds(g0, g1, g2, g3, 0);
#endif
        __builtin_amdgcn_s_wait_tensorcnt(0);
    }
#else
    { // cooperative fallback copy (16 x KCAT bf16 = 2*KCAT 16B chunks)
        const Q16* src = (const Q16*)(hcat + (long)m0 * KCAT);
        Q16* dst = (Q16*)lds;
        for (int i = threadIdx.x; i < 2 * KCAT; i += 256) dst[i] = src[i];
    }
#endif
    __syncthreads();

    v8f ai = {}, af = {}, ag = {}, ao = {};
    const int row  = lane & 15;
    const int ksel = (lane >> 4) << 3;      // 0 or 8 (A-fragment K sub-select)
    const int nt   = n0 >> 4;               // hidden n-tile (0..63)
    const int KT   = KCAT >> 5;

    for (int kk = 0; kk < KT; ++kk) {
        Frag fa;
        const unsigned short* ap = lds + row * KCAT + kk * 32 + ksel;
        fa.q[0] = *(const Q16*)ap;           // K  .. K+7
        fa.q[1] = *(const Q16*)(ap + 16);    // K+16 .. K+23
        const unsigned short* wp = wpack + (long)kk * (256 * 512) + lane * 16;
        #pragma unroll
        for (int g = 0; g < 4; ++g) {
            Frag fb;
            const unsigned short* bp = wp + (long)(g * 64 + nt) * 512;
            fb.q[0] = *(const Q16*)bp;
            fb.q[1] = *(const Q16*)(bp + 8);
            v8f& acc = (g == 0) ? ai : (g == 1) ? af : (g == 2) ? ag : ao;
            acc = __builtin_amdgcn_wmma_f32_16x16x32_bf16(false, fa.v, false, fb.v,
                                                          (short)0, acc, false, false);
        }
    }

    // ----- LSTM cell, fully in registers (gate order i,f,g,o) -----
    const int nglob = n0 + (lane & 15);
    #pragma unroll
    for (int r = 0; r < 8; ++r) {
        int m = m0 + r + ksel;               // C/D layout: lane<16 -> M=r, else M=r+8
        float gi = ai[r] + bias[nglob];
        float gf = af[r] + bias[1024 + nglob];
        float gg = ag[r] + bias[2048 + nglob];
        float go = ao[r] + bias[3072 + nglob];
        long cidx = (long)m * 1024 + nglob;
        float cold = c[cidx];
        float cnew = sigm(gf) * cold + sigm(gi) * tanhf(gg);
        float h = sigm(go) * tanhf(cnew);
        c[cidx] = cnew;
        unsigned short hb = f2bf(h);
        dstA[(long)m * strideA + offA + nglob] = hb;
        if (dstB) dstB[(long)m * strideB + offB + nglob] = hb;
    }
}

// ---------------------------------------------------------------------------
// out = h2 @ W_lin^T + b_lin  ([256x1024] x [1024x64]) via WMMA.
// grid = 16 (M tiles), block = 128 (4 waves = 4 N tiles).
// ---------------------------------------------------------------------------
__global__ void __launch_bounds__(128)
out_proj_kernel(const unsigned short* __restrict__ hcat2,
                const unsigned short* __restrict__ wlin,
                const float* __restrict__ blin,
                float* __restrict__ dout, int t, int TT) {
    const int lane = threadIdx.x & 31;
    const int nt   = threadIdx.x >> 5;       // 0..3
    const int m0   = blockIdx.x * 16;
    const int row  = lane & 15;
    const int ksel = (lane >> 4) << 3;
    v8f acc = {};
    for (int kk = 0; kk < 32; ++kk) {        // K = 1024
        Frag fa, fb;
        const unsigned short* ap = hcat2 + (long)(m0 + row) * 2048 + 1024 + kk * 32 + ksel;
        fa.q[0] = *(const Q16*)ap;
        fa.q[1] = *(const Q16*)(ap + 16);
        const unsigned short* bp = wlin + ((long)(kk * 4 + nt) * 32 + lane) * 16;
        fb.q[0] = *(const Q16*)bp;
        fb.q[1] = *(const Q16*)(bp + 8);
        acc = __builtin_amdgcn_wmma_f32_16x16x32_bf16(false, fa.v, false, fb.v,
                                                      (short)0, acc, false, false);
    }
    const int n = nt * 16 + (lane & 15);
    #pragma unroll
    for (int r = 0; r < 8; ++r) {
        int m = m0 + r + ksel;
        dout[((long)m * TT + t) * 64 + n] = acc[r] + blin[n];
    }
}

// ---------------------------------------------------------------------------
extern "C" void kernel_launch(void* const* d_in, const int* in_sizes, int n_in,
                              void* d_out, int out_size, void* d_ws, size_t ws_size,
                              hipStream_t stream) {
    const float* input = (const float*)d_in[0];
    const float* W_ih1 = (const float*)d_in[1];
    const float* W_hh1 = (const float*)d_in[2];
    const float* b_ih1 = (const float*)d_in[3];
    const float* b_hh1 = (const float*)d_in[4];
    const float* W_ih2 = (const float*)d_in[5];
    const float* W_hh2 = (const float*)d_in[6];
    const float* b_ih2 = (const float*)d_in[7];
    const float* b_hh2 = (const float*)d_in[8];
    const float* W_lin = (const float*)d_in[9];
    const float* b_lin = (const float*)d_in[10];
    const float* W_o2i = (const float*)d_in[11];
    const float* b_o2i = (const float*)d_in[12];
    float* dout = (float*)d_out;

    const int B = 256, IN = 64, OUT = 64;
    const int K1 = 1088, K2 = 2048;          // [x|h1], [h1|h2] K-concat
    int T  = in_sizes[0] / (B * IN);         // 256
    int TT = out_size / (B * OUT);           // 288
    int FUT = TT - T;                        // 32

    // ---- carve workspace ----
    char* ws = (char*)d_ws;
    size_t off = 0;
    auto carve = [&](size_t bytes) -> char* {
        char* p = ws + off;
        off = (off + bytes + 255) & ~(size_t)255;
        return p;
    };
    size_t szp1 = (size_t)(K1 / 32) * 256 * 512 * 2;   // 8.9 MB
    size_t szp2 = (size_t)(K2 / 32) * 256 * 512 * 2;   // 16.8 MB
    size_t szpl = (size_t)32 * 4 * 512 * 2;            // 128 KB
    unsigned short* pack1 = (unsigned short*)carve(szp1);
    unsigned short* pack2 = (unsigned short*)carve(szp2);
    unsigned short* packl = (unsigned short*)carve(szpl);
    float* bias1 = (float*)carve(4096 * 4);
    float* bias2 = (float*)carve(4096 * 4);
    unsigned short* hcat1a = (unsigned short*)carve((size_t)B * K1 * 2);
    unsigned short* hcat1b = (unsigned short*)carve((size_t)B * K1 * 2);
    unsigned short* hcat2a = (unsigned short*)carve((size_t)B * K2 * 2);
    unsigned short* hcat2b = (unsigned short*)carve((size_t)B * K2 * 2);
    float* c1 = (float*)carve((size_t)B * 1024 * 4);
    float* c2 = (float*)carve((size_t)B * 1024 * 4);

    // ---- one-time-per-call prep (idempotent, deterministic) ----
    {
        long tot1 = (long)(K1 / 32) * 256 * 512;
        pack_w_kernel<<<(unsigned)((tot1 + 255) / 256), 256, 0, stream>>>(
            W_ih1, W_hh1, 64, 1024, 256, pack1, tot1);
        long tot2 = (long)(K2 / 32) * 256 * 512;
        pack_w_kernel<<<(unsigned)((tot2 + 255) / 256), 256, 0, stream>>>(
            W_ih2, W_hh2, 1024, 1024, 256, pack2, tot2);
        long totl = (long)32 * 4 * 512;
        pack_w_kernel<<<(unsigned)((totl + 255) / 256), 256, 0, stream>>>(
            W_lin, W_lin, 1024, 1024, 4, packl, totl);
        bias_sum_kernel<<<16, 256, 0, stream>>>(b_ih1, b_hh1, bias1, 4096);
        bias_sum_kernel<<<16, 256, 0, stream>>>(b_ih2, b_hh2, bias2, 4096);
        hipMemsetAsync(hcat1a, 0, (size_t)B * K1 * 2, stream);
        hipMemsetAsync(hcat1b, 0, (size_t)B * K1 * 2, stream);
        hipMemsetAsync(hcat2a, 0, (size_t)B * K2 * 2, stream);
        hipMemsetAsync(hcat2b, 0, (size_t)B * K2 * 2, stream);
        hipMemsetAsync(c1, 0, (size_t)B * 1024 * 4, stream);
        hipMemsetAsync(c2, 0, (size_t)B * 1024 * 4, stream);
    }

    // ---- sequential scan; double-buffered activations avoid intra-step races ----
    dim3 gstep(B / 16, 1024 / 128);          // (16, 8)
    for (int t = 0; t < T + FUT; ++t) {
        int p = t & 1;
        unsigned short* h1r = p ? hcat1b : hcat1a;   // read buf: [x_t | h1_{t-1}]
        unsigned short* h1w = p ? hcat1a : hcat1b;   // write h1_t here (cols 64..)
        unsigned short* h2r = p ? hcat2b : hcat2a;   // read buf: [h1_t | h2_{t-1}]
        unsigned short* h2w = p ? hcat2a : hcat2b;   // write h2_t here (cols 1024..)

        if (t < T)
            prep_x_kernel<<<64, 256, 0, stream>>>(input, h1r, t, T);
        else
            proj_in_kernel<<<64, 256, 0, stream>>>(dout, W_o2i, b_o2i, h1r, t - 1, TT);

        lstm_step_kernel<<<gstep, 256, 16 * K1 * 2, stream>>>(
            h1r, K1, pack1, bias1, c1,
            h1w, K1, 64,           // h1 -> next step's layer-1 A
            h2r, K2, 0);           // h1 -> this step's layer-2 A
        lstm_step_kernel<<<gstep, 256, 16 * K2 * 2, stream>>>(
            h2r, K2, pack2, bias2, c2,
            h2w, K2, 1024,         // h2 -> next step's layer-2 A
            (unsigned short*)nullptr, 0, 0);
        out_proj_kernel<<<16, 128, 0, stream>>>(h2w, packl, b_lin, dout, t, TT);
    }
}
